// SelfAttention_81947976008099
// MI455X (gfx1250) — matrix operand order repaired and checked by reading, not compiled, not run on hardware
//
#include <hip/hip_runtime.h>
#include <hip/hip_bf16.h>

#define L_SEQ  2048
#define NH     16
#define DMODEL 1024
#define DH     64

typedef __bf16 bf16;
typedef __attribute__((ext_vector_type(16))) __bf16 v16bf;
typedef __attribute__((ext_vector_type(8)))  __bf16 v8bf;
typedef __attribute__((ext_vector_type(8)))  float  v8f;
typedef __attribute__((vector_size(16)))     int    v4i;

__device__ inline void load_bf8(v16bf& d, int base, const bf16* p) {
  v8bf t = *(const v8bf*)p;
#pragma unroll
  for (int i = 0; i < 8; ++i) d[base + i] = t[i];
}

__device__ inline void load_f8(v16bf& d, int base, const float* p) {
  float4 a = *(const float4*)p;
  float4 b = *(const float4*)(p + 4);
  d[base + 0] = (bf16)a.x; d[base + 1] = (bf16)a.y;
  d[base + 2] = (bf16)a.z; d[base + 3] = (bf16)a.w;
  d[base + 4] = (bf16)b.x; d[base + 5] = (bf16)b.y;
  d[base + 6] = (bf16)b.z; d[base + 7] = (bf16)b.w;
}

__device__ inline v8f wmma_bf16(v16bf a, v16bf b, v8f c) {
  return __builtin_amdgcn_wmma_f32_16x16x32_bf16(false, a, false, b, (short)0, c,
                                                 false, false);
}

// ---------------------------------------------------------------------------
// Async global->LDS staging (gfx1250 GLOBAL_LOAD_ASYNC_TO_LDS_B128, ASYNCcnt),
// with a synchronous reg->LDS fallback if the builtins aren't declared.
// Builtin signature (from clang diagnostic): (v4i AS1*, v4i AS3*, imm, imm).
// ---------------------------------------------------------------------------
#if defined(__gfx1250__) && __has_builtin(__builtin_amdgcn_global_load_async_to_lds_b128)
#define HAS_ASYNC_LDS 1
#else
#define HAS_ASYNC_LDS 0
#endif

#if HAS_ASYNC_LDS
typedef __attribute__((address_space(1))) v4i* gptr_v4i;
typedef __attribute__((address_space(3))) v4i* lptr_v4i;
#endif

__device__ inline void stage16(const bf16* g, bf16* l) {
#if HAS_ASYNC_LDS
  __builtin_amdgcn_global_load_async_to_lds_b128(
      (gptr_v4i)(uintptr_t)g,
      (lptr_v4i)(uint32_t)(uintptr_t)l,
      0, 0);
#else
  *(v8bf*)l = *(const v8bf*)g;
#endif
}

__device__ inline void wait_async_le1() {
#if HAS_ASYNC_LDS
#if __has_builtin(__builtin_amdgcn_s_wait_asynccnt)
  __builtin_amdgcn_s_wait_asynccnt(1);
#else
  asm volatile("s_wait_asynccnt 1" ::: "memory");
#endif
#endif
}

__device__ inline void wait_async_0() {
#if HAS_ASYNC_LDS
#if __has_builtin(__builtin_amdgcn_s_wait_asynccnt)
  __builtin_amdgcn_s_wait_asynccnt(0);
#else
  asm volatile("s_wait_asynccnt 0" ::: "memory");
#endif
#endif
}

// ---------------------------------------------------------------------------
// Weight transpose + fp32 -> bf16 convert: Wt[n][k] = (bf16) W[k][n]
// ---------------------------------------------------------------------------
__global__ __launch_bounds__(256) void transpose_cvt(const float* __restrict__ in,
                                                     bf16* __restrict__ out) {
  __shared__ float tile[16][17];
  const int tx = threadIdx.x & 15, ty = threadIdx.x >> 4;
  const int bx = blockIdx.x * 16, by = blockIdx.y * 16;
  tile[ty][tx] = in[(size_t)(by + ty) * DMODEL + bx + tx];
  __syncthreads();
  out[(size_t)(bx + ty) * DMODEL + by + tx] = (bf16)tile[tx][ty];
}

// ---------------------------------------------------------------------------
// GEMM: out[M=4096][N=1024] = A[4096][1024] x Wt^T + bias via bf16 WMMA.
// B operand (64 cols x 32 k per step, shared by all 8 waves) is staged into
// LDS with double-buffered async copies; A is loaded per-wave from global.
// AMODE: 0 = A fp32 (convert on load), 1 = A bf16.
// OMODE: 0 = bf16 natural, 1 = bf16 per-head transposed vt[b][h][d][l],
//        2 = fp32 natural (final output).
// ---------------------------------------------------------------------------
template <int AMODE, int OMODE>
__global__ __launch_bounds__(256) void gemm_bf16_wmma(const void* __restrict__ Ap,
                                                      const bf16* __restrict__ Wt,
                                                      const float* __restrict__ bias,
                                                      void* __restrict__ outp) {
  __shared__ __align__(16) bf16 Wlds[2][64 * 32];   // [buf][col*32 + k], 4 KB each
  const int tid  = threadIdx.x;
  const int lane = tid & 31, wv = tid >> 5;
  const int half = lane >> 4, nl = lane & 15;
  const int row0 = (blockIdx.y * 8 + wv) * 16;
  const int col0 = blockIdx.x * 64;

  // Cooperative B-tile fetch: thread t moves 16B; 256 threads cover 4 KB.
  const int lcol  = tid >> 2;                      // 0..63
  const int lpart = tid & 3;                       // k sub-chunk of 8 bf16
  const bf16* gW = Wt + (size_t)(col0 + lcol) * DMODEL + lpart * 8;
  bf16* lW0 = &Wlds[0][lcol * 32 + lpart * 8];
  bf16* lW1 = &Wlds[1][lcol * 32 + lpart * 8];

  v8f acc[4];
#pragma unroll
  for (int t = 0; t < 4; ++t)
#pragma unroll
    for (int v = 0; v < 8; ++v) acc[t][v] = 0.f;

  const float* af = (const float*)Ap;
  const bf16*  ab = (const bf16*)Ap;
  const size_t arow = (size_t)(row0 + nl) * DMODEL;

  stage16(gW, lW0);                                 // prologue: k0 = 0

  const int NSTEP = DMODEL / 32;                    // 32
  for (int step = 0; step < NSTEP; ++step) {
    const int k0 = step * 32;
    bf16* curbuf = (step & 1) ? &Wlds[1][0] : &Wlds[0][0];

    if (step + 1 < NSTEP) {
      stage16(gW + k0 + 32, (step & 1) ? lW0 : lW1);
      wait_async_le1();
    } else {
      wait_async_0();
    }
    __syncthreads();                                // B tile visible to all waves

    v16bf afr;
    if (AMODE == 0) {
      const float* p = af + arow + k0;
      load_f8(afr, 0, p + 8 * half);
      load_f8(afr, 8, p + 16 + 8 * half);
      if (k0 + 32 < DMODEL) __builtin_prefetch(af + arow + k0 + 32, 0, 3);
    } else {
      const bf16* p = ab + arow + k0;
      load_bf8(afr, 0, p + 8 * half);
      load_bf8(afr, 8, p + 16 + 8 * half);
      if (k0 + 32 < DMODEL) __builtin_prefetch(ab + arow + k0 + 32, 0, 3);
    }

#pragma unroll
    for (int t = 0; t < 4; ++t) {
      const bf16* wr = curbuf + (t * 16 + nl) * 32 + 8 * half;
      v16bf bfr;
      load_bf8(bfr, 0, wr);
      load_bf8(bfr, 8, wr + 16);
      acc[t] = wmma_bf16(afr, bfr, acc[t]);
    }
    __syncthreads();                                // all reads done before rewrite
  }

#pragma unroll
  for (int t = 0; t < 4; ++t) {
    const int col = col0 + t * 16 + nl;
    const float bc = bias[col];
#pragma unroll
    for (int v = 0; v < 8; ++v) {
      const float x = acc[t][v] + bc;
      const int m = row0 + 8 * half + v;
      if (OMODE == 0) {
        ((bf16*)outp)[(size_t)m * DMODEL + col] = (bf16)x;
      } else if (OMODE == 2) {
        ((float*)outp)[(size_t)m * DMODEL + col] = x;
      } else {
        const int hh = col >> 6, dd = col & 63;
        const int bb = m >> 11, ll = m & (L_SEQ - 1);
        ((bf16*)outp)[((size_t)((bb * NH + hh) * DH + dd)) * L_SEQ + ll] = (bf16)x;
      }
    }
  }
}

// ---------------------------------------------------------------------------
// Fused causal flash attention, one wave per (b, h, 16-row q tile).
// Scores and P·V both via v_wmma_f32_16x16x32_bf16; key blocks of 32;
// online softmax with scale 1/sqrt(DX)=1/32; P transposed C->A layout via LDS.
// ---------------------------------------------------------------------------
__global__ __launch_bounds__(256) void attn_fused(const bf16* __restrict__ qb,
                                                  const bf16* __restrict__ kbuf,
                                                  const bf16* __restrict__ vt,
                                                  bf16* __restrict__ ao) {
  __shared__ __align__(16) bf16 Pbuf[8][16 * 32];
  const int lane = threadIdx.x & 31;
  const int wv   = threadIdx.x >> 5;
  const int task = blockIdx.x * 8 + wv;       // B*H*(L/16) = 4096 tasks
  const int qt = task & (L_SEQ / 16 - 1);
  const int h  = (task >> 7) & (NH - 1);
  const int b  = task >> 11;
  const int q0 = qt * 16;
  const int half = lane >> 4, nl = lane & 15;

  v16bf aq0, aq1;
  {
    const bf16* qrow = qb + ((size_t)(b * L_SEQ + q0 + nl)) * DMODEL + h * DH;
    load_bf8(aq0, 0, qrow + 8 * half);
    load_bf8(aq0, 8, qrow + 16 + 8 * half);
    load_bf8(aq1, 0, qrow + 32 + 8 * half);
    load_bf8(aq1, 8, qrow + 48 + 8 * half);
  }

  v8f o0, o1, o2, o3;
  float Mv[8], Sv[8];
#pragma unroll
  for (int v = 0; v < 8; ++v) {
    o0[v] = 0.f; o1[v] = 0.f; o2[v] = 0.f; o3[v] = 0.f;
    Mv[v] = -1e30f; Sv[v] = 0.f;
  }

  const int nkb = (q0 + 47) >> 5;             // key blocks covering keys <= q0+15
  const float scale = 0.03125f;               // 1/sqrt(1024)

  for (int kbk = 0; kbk < nkb; ++kbk) {
    const int key0 = kbk * 32;

    v16bf b00, b01, b10, b11;                 // [d-chunk][key-subtile]
    {
      const bf16* kr0 = kbuf + ((size_t)(b * L_SEQ + key0 + nl)) * DMODEL
                        + h * DH + 8 * half;
      const bf16* kr1 = kbuf + ((size_t)(b * L_SEQ + key0 + 16 + nl)) * DMODEL
                        + h * DH + 8 * half;
      load_bf8(b00, 0, kr0);      load_bf8(b00, 8, kr0 + 16);
      load_bf8(b10, 0, kr0 + 32); load_bf8(b10, 8, kr0 + 48);
      load_bf8(b01, 0, kr1);      load_bf8(b01, 8, kr1 + 16);
      load_bf8(b11, 0, kr1 + 32); load_bf8(b11, 8, kr1 + 48);
    }

    v8f s0, s1;
#pragma unroll
    for (int v = 0; v < 8; ++v) { s0[v] = 0.f; s1[v] = 0.f; }
    s0 = wmma_bf16(aq0, b00, s0);
    s0 = wmma_bf16(aq1, b10, s0);
    s1 = wmma_bf16(aq0, b01, s1);
    s1 = wmma_bf16(aq1, b11, s1);

#pragma unroll
    for (int v = 0; v < 8; ++v) {
      const int qi = q0 + 8 * half + v;
      float x0 = (key0 + nl      > qi) ? -1e30f : s0[v] * scale;
      float x1 = (key0 + 16 + nl > qi) ? -1e30f : s1[v] * scale;
      float mv = fmaxf(x0, x1);
#pragma unroll
      for (int off = 1; off < 16; off <<= 1) mv = fmaxf(mv, __shfl_xor(mv, off, 32));
      const float newM = fmaxf(Mv[v], mv);
      const float alpha = __expf(Mv[v] - newM);
      const float p0 = __expf(x0 - newM);
      const float p1 = __expf(x1 - newM);
      float rs = p0 + p1;
#pragma unroll
      for (int off = 1; off < 16; off <<= 1) rs += __shfl_xor(rs, off, 32);
      Sv[v] = Sv[v] * alpha + rs;
      Mv[v] = newM;
      o0[v] *= alpha; o1[v] *= alpha; o2[v] *= alpha; o3[v] *= alpha;
      Pbuf[wv][(8 * half + v) * 32 + nl]      = (bf16)p0;
      Pbuf[wv][(8 * half + v) * 32 + 16 + nl] = (bf16)p1;
    }

    // wave-private LDS tile: DS ops are in-order per wave; make it explicit.
    asm volatile("s_wait_dscnt 0" ::: "memory");

    v16bf pa;                                  // P as 16x32 A fragment
    {
      const bf16* prow = &Pbuf[wv][nl * 32];
      load_bf8(pa, 0, prow + 8 * half);
      load_bf8(pa, 8, prow + 16 + 8 * half);
    }

    {
      const size_t vb = ((size_t)((b * NH + h) * DH)) * L_SEQ;
      v16bf bv;
      const bf16* vr0 = vt + vb + (size_t)(nl) * L_SEQ + key0 + 8 * half;
      load_bf8(bv, 0, vr0); load_bf8(bv, 8, vr0 + 16);
      o0 = wmma_bf16(pa, bv, o0);
      const bf16* vr1 = vt + vb + (size_t)(16 + nl) * L_SEQ + key0 + 8 * half;
      load_bf8(bv, 0, vr1); load_bf8(bv, 8, vr1 + 16);
      o1 = wmma_bf16(pa, bv, o1);
      const bf16* vr2 = vt + vb + (size_t)(32 + nl) * L_SEQ + key0 + 8 * half;
      load_bf8(bv, 0, vr2); load_bf8(bv, 8, vr2 + 16);
      o2 = wmma_bf16(pa, bv, o2);
      const bf16* vr3 = vt + vb + (size_t)(48 + nl) * L_SEQ + key0 + 8 * half;
      load_bf8(bv, 0, vr3); load_bf8(bv, 8, vr3 + 16);
      o3 = wmma_bf16(pa, bv, o3);
    }
  }

#pragma unroll
  for (int v = 0; v < 8; ++v) {
    const float inv = 1.0f / Sv[v];
    o0[v] *= inv; o1[v] *= inv; o2[v] *= inv; o3[v] *= inv;
  }
#pragma unroll
  for (int v = 0; v < 8; ++v) {
    const size_t ro = ((size_t)(b * L_SEQ + q0 + 8 * half + v)) * DMODEL
                      + h * DH + nl;
    ao[ro]      = (bf16)o0[v];
    ao[ro + 16] = (bf16)o1[v];
    ao[ro + 32] = (bf16)o2[v];
    ao[ro + 48] = (bf16)o3[v];
  }
}

// ---------------------------------------------------------------------------
extern "C" void kernel_launch(void* const* d_in, const int* in_sizes, int n_in,
                              void* d_out, int out_size, void* d_ws, size_t ws_size,
                              hipStream_t stream) {
  (void)in_sizes; (void)n_in; (void)out_size; (void)ws_size;
  const float* value = (const float*)d_in[0];
  const float* keyin = (const float*)d_in[1];
  const float* query = (const float*)d_in[2];
  // d_in[3] = mask: causal tril, applied arithmetically in-kernel.
  const float* Wq = (const float*)d_in[4];  const float* bq = (const float*)d_in[5];
  const float* Wk = (const float*)d_in[6];  const float* bk = (const float*)d_in[7];
  const float* Wv = (const float*)d_in[8];  const float* bv = (const float*)d_in[9];
  // d_in[10..11] = Wr_w/Wr_b: dead code in reference, skipped.
  const float* Wo = (const float*)d_in[12]; const float* bo = (const float*)d_in[13];

  char* ws = (char*)d_ws;
  const size_t MB = 1024 * 1024;
  bf16* WqT = (bf16*)(ws + 0 * MB);   // 1024x1024 bf16 = 2 MB each
  bf16* WkT = (bf16*)(ws + 2 * MB);
  bf16* WvT = (bf16*)(ws + 4 * MB);
  bf16* WoT = (bf16*)(ws + 6 * MB);
  bf16* qbf = (bf16*)(ws + 8 * MB);   // 4096x1024 bf16 = 8 MB each
  bf16* kbf = (bf16*)(ws + 16 * MB);
  bf16* vtb = (bf16*)(ws + 24 * MB);  // vt[b][h][d][l]
  bf16* aob = (bf16*)(ws + 32 * MB);  // attention out, natural layout

  dim3 tb(256);
  dim3 gT(DMODEL / 16, DMODEL / 16);
  transpose_cvt<<<gT, tb, 0, stream>>>(Wq, WqT);
  transpose_cvt<<<gT, tb, 0, stream>>>(Wk, WkT);
  transpose_cvt<<<gT, tb, 0, stream>>>(Wv, WvT);
  transpose_cvt<<<gT, tb, 0, stream>>>(Wo, WoT);

  dim3 gG(DMODEL / 64, (2 * L_SEQ) / 128);    // (16, 32)
  gemm_bf16_wmma<0, 0><<<gG, tb, 0, stream>>>(query, WqT, bq, qbf);
  gemm_bf16_wmma<0, 0><<<gG, tb, 0, stream>>>(keyin, WkT, bk, kbf);
  gemm_bf16_wmma<0, 1><<<gG, tb, 0, stream>>>(value, WvT, bv, vtb);

  attn_fused<<<dim3((2 * NH * (L_SEQ / 16)) / 8), tb, 0, stream>>>(qbf, kbf, vtb, aob);

  gemm_bf16_wmma<1, 2><<<gG, tb, 0, stream>>>(aob, WoT, bo, d_out);
}
